// GraphSAGEFraudDetector_29772713296000
// MI455X (gfx1250) — matrix operand order repaired and checked by reading, not compiled, not run on hardware
//
#include <hip/hip_runtime.h>
#include <hip/hip_bf16.h>

// ---------------------------------------------------------------------------
// Hetero GraphSAGE fraud detector for MI455X (gfx1250, wave32, WMMA bf16)
// Weights pre-packed once per call into bf16 WMMA B-fragment layout; GEMM
// waves compute 16 x (NT*16) strips with NT independent accumulators.
// ---------------------------------------------------------------------------

typedef __attribute__((ext_vector_type(16))) __bf16 v16bf;
typedef __attribute__((ext_vector_type(8)))  float  v8f;

#define N_ACC 200000
#define N_MER 100000
#define D_ACC 64
#define D_MER 32
#define HDIM  128
#define OUTD  64
#define CLFD  64

__device__ __forceinline__ float warpSum32(float v) {
#pragma unroll
    for (int o = 16; o > 0; o >>= 1) v += __shfl_xor(v, o, 32);
    return v;
}

__device__ __forceinline__ void atomAddF(float* p, float v) {
    unsafeAtomicAdd(p, v);   // hardware global_atomic_add_f32
}

// A-fragment (16x32 bf16, MxK): lane<16 -> K = {0..7,16..23}, lane>=16 -> {8..15,24..31}
// p points at &A[row*K + kb + hlf*8]
__device__ __forceinline__ v16bf load_frag_a(const float* __restrict__ p) {
    float4 x0 = *(const float4*)(p);
    float4 x1 = *(const float4*)(p + 4);
    float4 x2 = *(const float4*)(p + 16);
    float4 x3 = *(const float4*)(p + 20);
    v16bf a;
    a[0]  = (__bf16)x0.x; a[1]  = (__bf16)x0.y; a[2]  = (__bf16)x0.z; a[3]  = (__bf16)x0.w;
    a[4]  = (__bf16)x1.x; a[5]  = (__bf16)x1.y; a[6]  = (__bf16)x1.z; a[7]  = (__bf16)x1.w;
    a[8]  = (__bf16)x2.x; a[9]  = (__bf16)x2.y; a[10] = (__bf16)x2.z; a[11] = (__bf16)x2.w;
    a[12] = (__bf16)x3.x; a[13] = (__bf16)x3.y; a[14] = (__bf16)x3.z; a[15] = (__bf16)x3.w;
    return a;
}

// Repack W[N,K] (fp32, row-major, N x K) into bf16 WMMA B-fragments:
// WB[((nt*(K/32) + kb)*32 + lane)*16 + e] = W[(nt*16 + (lane&15))*K + kb*32 + (lane>>4)*8 + (e>>3)*16 + (e&7)]
__global__ __launch_bounds__(256) void pack_w_kernel(
    const float* __restrict__ W, __bf16* __restrict__ WB, int N, int K)
{
    int i = blockIdx.x * blockDim.x + threadIdx.x;
    if (i >= N * K) return;
    int KB   = K >> 5;
    int e    = i & 15;
    int lane = (i >> 4) & 31;
    int frag = i >> 9;
    int kb   = frag % KB;
    int nt   = frag / KB;
    int col  = nt * 16 + (lane & 15);
    int k    = kb * 32 + (lane >> 4) * 8 + (e >> 3) * 16 + (e & 7);
    WB[i] = (__bf16)W[(size_t)col * K + k];
}

// Out[M, NT*16] = A1[M,K] @ W1^T (+ A2[M,K] @ W2^T) + bias, optional ReLU.
// W1/W2 are pre-packed bf16 fragment arrays. One wave: 16 rows x full N.
template <int NT>
__global__ __launch_bounds__(256) void gemm_wmma_kernel(
    const float* __restrict__ A1, const __bf16* __restrict__ W1,
    const float* __restrict__ A2, const __bf16* __restrict__ W2,
    const float* __restrict__ bias, float* __restrict__ Out,
    int M, int K, int doRelu)
{
    constexpr int N = NT * 16;
    int wv = blockIdx.x * 8 + (threadIdx.x >> 5);
    if (wv >= (M >> 4)) return;
    int lane = threadIdx.x & 31;
    int hlf  = lane >> 4;
    int l    = lane & 15;
    int row  = (wv << 4) + l;   // A row for this lane (both halves)
    int KB   = K >> 5;

    v8f acc[NT];
#pragma unroll
    for (int nt = 0; nt < NT; ++nt) {
        float bv = bias[nt * 16 + l];   // C layout: N = lane&15 within tile
#pragma unroll
        for (int r = 0; r < 8; ++r) acc[nt][r] = bv;
    }

    const float*  pa = A1 + (size_t)row * K + hlf * 8;
    const __bf16* pb = W1 + (size_t)lane * 16;
    for (int kb = 0; kb < KB; ++kb) {
        v16bf a = load_frag_a(pa + kb * 32);
#pragma unroll
        for (int nt = 0; nt < NT; ++nt) {
            v16bf b = *(const v16bf*)(pb + ((size_t)(nt * KB + kb) << 9));
            acc[nt] = __builtin_amdgcn_wmma_f32_16x16x32_bf16(false, a, false, b,
                                                              (short)0, acc[nt],
                                                              false, false);
        }
    }
    if (A2 != nullptr) {
        const float*  pa2 = A2 + (size_t)row * K + hlf * 8;
        const __bf16* pb2 = W2 + (size_t)lane * 16;
        for (int kb = 0; kb < KB; ++kb) {
            v16bf a = load_frag_a(pa2 + kb * 32);
#pragma unroll
            for (int nt = 0; nt < NT; ++nt) {
                v16bf b = *(const v16bf*)(pb2 + ((size_t)(nt * KB + kb) << 9));
                acc[nt] = __builtin_amdgcn_wmma_f32_16x16x32_bf16(false, a, false, b,
                                                                  (short)0, acc[nt],
                                                                  false, false);
            }
        }
    }
    // C/D layout: VGPR r, lanes 0-15 -> M=r, lanes 16-31 -> M=8+r; N = lane&15
    int rb = (wv << 4) + (hlf << 3);
#pragma unroll
    for (int nt = 0; nt < NT; ++nt) {
#pragma unroll
        for (int r = 0; r < 8; ++r) {
            float v = acc[nt][r];
            if (doRelu) v = fmaxf(v, 0.0f);
            Out[(size_t)(rb + r) * N + nt * 16 + l] = v;
        }
    }
}

// One wave per edge: lane handles 4 contiguous floats of the 128-wide row.
__global__ __launch_bounds__(256) void scatter_add_kernel(
    const float* __restrict__ X, const int* __restrict__ src, const int* __restrict__ dst,
    int nE, float* __restrict__ Agg, float* __restrict__ Cnt)
{
    int e = blockIdx.x * 8 + (threadIdx.x >> 5);
    if (e >= nE) return;
    int lane = threadIdx.x & 31;
    int s = src[e], d = dst[e];
    float4 v = *(const float4*)(X + (size_t)s * HDIM + lane * 4);
    float* o = Agg + (size_t)d * HDIM + lane * 4;
    atomAddF(o + 0, v.x);
    atomAddF(o + 1, v.y);
    atomAddF(o + 2, v.z);
    atomAddF(o + 3, v.w);
    if (lane == 0) atomAddF(Cnt + d, 1.0f);
}

__global__ __launch_bounds__(256) void mean_div_kernel(
    float* __restrict__ Agg, const float* __restrict__ Cnt, int total, int shift)
{
    int i = blockIdx.x * blockDim.x + threadIdx.x;
    if (i >= total) return;
    float c = Cnt[i >> shift];
    Agg[i] = Agg[i] / fmaxf(c, 1.0f);
}

// P = P/max(||P||,1e-12) (+ Q/max(||Q||,1e-12) when Q != nullptr), per row. One wave/row.
__global__ __launch_bounds__(256) void l2norm_kernel(
    float* __restrict__ P, const float* __restrict__ Q, int M, int width)
{
    int row = blockIdx.x * 8 + (threadIdx.x >> 5);
    if (row >= M) return;
    int lane = threadIdx.x & 31;
    int n = width >> 5;   // 4 (width 128) or 2 (width 64)
    float vp[4], vq[4];
    float sp = 0.0f, sq = 0.0f;
#pragma unroll 4
    for (int j = 0; j < n; ++j) {
        vp[j] = P[(size_t)row * width + j * 32 + lane];
        sp += vp[j] * vp[j];
    }
    if (Q != nullptr) {
#pragma unroll 4
        for (int j = 0; j < n; ++j) {
            vq[j] = Q[(size_t)row * width + j * 32 + lane];
            sq += vq[j] * vq[j];
        }
    }
    sp = warpSum32(sp);
    float ip = 1.0f / fmaxf(sqrtf(sp), 1e-12f);
    float iq = 0.0f;
    if (Q != nullptr) {
        sq = warpSum32(sq);
        iq = 1.0f / fmaxf(sqrtf(sq), 1e-12f);
    }
#pragma unroll 4
    for (int j = 0; j < n; ++j) {
        float v = vp[j] * ip;
        if (Q != nullptr) v += vq[j] * iq;
        P[(size_t)row * width + j * 32 + lane] = v;
    }
}

// Column sums and sums-of-squares (width = 1<<shift, 64 or 128).
__global__ __launch_bounds__(256) void bn_stats_kernel(
    const float* __restrict__ X, int M, int shift,
    float* __restrict__ Sum, float* __restrict__ Sq)
{
    int width = 1 << shift;
    int col = threadIdx.x & (width - 1);
    int sub = threadIdx.x >> shift;
    int rpb = 256 >> shift;
    int rstride = gridDim.x * rpb;
    float s = 0.0f, q = 0.0f;
    for (int r = blockIdx.x * rpb + sub; r < M; r += rstride) {
        float v = X[((size_t)r << shift) + col];
        s += v;
        q += v * v;
    }
    atomAddF(&Sum[col], s);
    atomAddF(&Sq[col], q);
}

__global__ __launch_bounds__(256) void bn_apply_kernel(
    const float* __restrict__ X, float* __restrict__ Out,
    const float* __restrict__ Sum, const float* __restrict__ Sq,
    const float* __restrict__ g, const float* __restrict__ b,
    int total, int shift, float invM, int doRelu)
{
    int i = blockIdx.x * blockDim.x + threadIdx.x;
    if (i >= total) return;
    int col = i & ((1 << shift) - 1);
    float mu  = Sum[col] * invM;
    float var = Sq[col] * invM - mu * mu;
    float sc  = g[col] * rsqrtf(var + 1e-5f);
    float v   = (X[i] - mu) * sc + b[col];
    if (doRelu) v = fmaxf(v, 0.0f);
    Out[i] = v;
}

// logits[i] = b2 + sum_j clfW2[j] * Hb[i][j], Hb width = 64. One wave per row.
__global__ __launch_bounds__(256) void logits_kernel(
    const float* __restrict__ Hb, const float* __restrict__ W2,
    const float* __restrict__ b2, float* __restrict__ Out, int M)
{
    int row = blockIdx.x * 8 + (threadIdx.x >> 5);
    if (row >= M) return;
    int lane = threadIdx.x & 31;
    const float* h = Hb + (size_t)row * CLFD;
    float s = h[lane] * W2[lane] + h[lane + 32] * W2[lane + 32];
    s = warpSum32(s);
    if (lane == 0) Out[row] = s + b2[0];
}

// ---------------------------------------------------------------------------

extern "C" void kernel_launch(void* const* d_in, const int* in_sizes, int n_in,
                              void* d_out, int out_size, void* d_ws, size_t ws_size,
                              hipStream_t stream)
{
    (void)n_in; (void)out_size; (void)ws_size;

    const float* x_acc   = (const float*)d_in[0];
    const float* x_mer   = (const float*)d_in[1];
    const int*   ei_pays = (const int*)d_in[2];
    const int*   ei_rev  = (const int*)d_in[3];
    const int*   ei_tr   = (const int*)d_in[4];
    const int E_pays = in_sizes[2] / 2;
    const int E_rev  = in_sizes[3] / 2;
    const int E_tr   = in_sizes[4] / 2;

    const float* projW_acc = (const float*)d_in[5];
    const float* projb_acc = (const float*)d_in[6];
    const float* projW_mer = (const float*)d_in[7];
    const float* projb_mer = (const float*)d_in[8];

    const float* Wl[3] = { (const float*)d_in[9],  (const float*)d_in[12], (const float*)d_in[15] };
    const float* bl[3] = { (const float*)d_in[10], (const float*)d_in[13], (const float*)d_in[16] };
    const float* Wr[3] = { (const float*)d_in[11], (const float*)d_in[14], (const float*)d_in[17] };
    const float* bng[3] = { (const float*)d_in[18], (const float*)d_in[20], (const float*)d_in[22] };
    const float* bnb[3] = { (const float*)d_in[19], (const float*)d_in[21], (const float*)d_in[23] };
    const float* clfW1 = (const float*)d_in[24];
    const float* clfb1 = (const float*)d_in[25];
    const float* clfW2 = (const float*)d_in[26];
    const float* clfb2 = (const float*)d_in[27];

    float* ws = (float*)d_ws;
    size_t o = 0;
    float* XA = ws + o; o += (size_t)N_ACC * HDIM;
    float* XM = ws + o; o += (size_t)N_MER * HDIM;
    float* G1 = ws + o; o += (size_t)N_MER * HDIM;   // agg for merchant (pays)
    float* G2 = ws + o; o += (size_t)N_ACC * HDIM;   // agg for account (rev_pays)
    float* G3 = ws + o; o += (size_t)N_ACC * HDIM;   // agg for account (transfer)
    float* P1 = ws + o; o += (size_t)N_MER * HDIM;   // pre-BN merchant / classifier hidden
    float* P2 = ws + o; o += (size_t)N_ACC * HDIM;
    float* P3 = ws + o; o += (size_t)N_ACC * HDIM;
    float* CM  = ws + o; o += N_MER;
    float* CA1 = ws + o; o += N_ACC;
    float* CA2 = ws + o; o += N_ACC;
    float* ST  = ws + o; o += 512;   // sumA[128], sqA[128], sumM[128], sqM[128]

    // bf16 packed-weight arena (32B-aligned: o is a multiple of 8 floats)
    __bf16* WSB = (__bf16*)(ws + o);
    size_t ob = 0;
    auto pack = [&](const float* W, int N, int K) -> const __bf16* {
        __bf16* dst = WSB + ob;
        ob += (size_t)N * K;
        int total = N * K;
        hipLaunchKernelGGL(pack_w_kernel, dim3((total + 255) / 256), dim3(256), 0, stream,
                           W, dst, N, K);
        return dst;
    };

    const int wouts[3] = { HDIM, HDIM, OUTD };

    const __bf16* PprojA = pack(projW_acc, HDIM, D_ACC);
    const __bf16* PprojM = pack(projW_mer, HDIM, D_MER);
    const __bf16* PWl[3][3];
    const __bf16* PWr[3][3];
    for (int l = 0; l < 3; ++l) {
        for (int t = 0; t < 3; ++t) {
            PWl[l][t] = pack(Wl[l] + (size_t)t * wouts[l] * HDIM, wouts[l], HDIM);
            PWr[l][t] = pack(Wr[l] + (size_t)t * wouts[l] * HDIM, wouts[l], HDIM);
        }
    }
    const __bf16* Pclf1 = pack(clfW1, CLFD, OUTD);

    // N must be 128 (NT=8) or 64 (NT=4)
    auto gemm = [&](const float* A1, const __bf16* W1, const float* A2, const __bf16* W2,
                    const float* bias, float* Out, int M, int N, int K, int relu) {
        int waves  = M >> 4;
        int blocks = (waves + 7) >> 3;
        if (N == 128) {
            hipLaunchKernelGGL(gemm_wmma_kernel<8>, dim3(blocks), dim3(256), 0, stream,
                               A1, W1, A2, W2, bias, Out, M, K, relu);
        } else {
            hipLaunchKernelGGL(gemm_wmma_kernel<4>, dim3(blocks), dim3(256), 0, stream,
                               A1, W1, A2, W2, bias, Out, M, K, relu);
        }
    };

    // ---- input projections: XA = x_acc @ projW_acc^T + b ; XM likewise ----
    gemm(x_acc, PprojA, nullptr, nullptr, projb_acc, XA, N_ACC, HDIM, D_ACC, 0);
    gemm(x_mer, PprojM, nullptr, nullptr, projb_mer, XM, N_MER, HDIM, D_MER, 0);

    for (int l = 0; l < 3; ++l) {
        const int wo = wouts[l];
        const int sh = (wo == 128) ? 7 : 6;

        hipMemsetAsync(G1, 0, (size_t)N_MER * HDIM * sizeof(float), stream);
        hipMemsetAsync(G2, 0, (size_t)N_ACC * HDIM * sizeof(float), stream);
        hipMemsetAsync(G3, 0, (size_t)N_ACC * HDIM * sizeof(float), stream);
        hipMemsetAsync(CM,  0, (size_t)N_MER * sizeof(float), stream);
        hipMemsetAsync(CA1, 0, (size_t)N_ACC * sizeof(float), stream);
        hipMemsetAsync(CA2, 0, (size_t)N_ACC * sizeof(float), stream);

        // scatter mean aggregation (sources are width-128 features)
        hipLaunchKernelGGL(scatter_add_kernel, dim3((E_pays + 7) / 8), dim3(256), 0, stream,
                           XA, ei_pays, ei_pays + E_pays, E_pays, G1, CM);
        hipLaunchKernelGGL(scatter_add_kernel, dim3((E_rev + 7) / 8), dim3(256), 0, stream,
                           XM, ei_rev, ei_rev + E_rev, E_rev, G2, CA1);
        hipLaunchKernelGGL(scatter_add_kernel, dim3((E_tr + 7) / 8), dim3(256), 0, stream,
                           XA, ei_tr, ei_tr + E_tr, E_tr, G3, CA2);

        hipLaunchKernelGGL(mean_div_kernel, dim3((N_MER * HDIM + 255) / 256), dim3(256), 0, stream,
                           G1, CM, N_MER * HDIM, 7);
        hipLaunchKernelGGL(mean_div_kernel, dim3((N_ACC * HDIM + 255) / 256), dim3(256), 0, stream,
                           G2, CA1, N_ACC * HDIM, 7);
        hipLaunchKernelGGL(mean_div_kernel, dim3((N_ACC * HDIM + 255) / 256), dim3(256), 0, stream,
                           G3, CA2, N_ACC * HDIM, 7);

        // SAGE linear parts: P = mean @ Wl^T + bl + x_dst @ Wr^T (dual-K WMMA GEMM)
        gemm(G1, PWl[l][0], XM, PWr[l][0], bl[l] + 0 * wo, P1, N_MER, wo, HDIM, 0);
        gemm(G2, PWl[l][1], XA, PWr[l][1], bl[l] + 1 * wo, P2, N_ACC, wo, HDIM, 0);
        gemm(G3, PWl[l][2], XA, PWr[l][2], bl[l] + 2 * wo, P3, N_ACC, wo, HDIM, 0);

        // per-SAGE L2 row normalization; account sums its two edge-type outputs
        hipLaunchKernelGGL(l2norm_kernel, dim3((N_ACC + 7) / 8), dim3(256), 0, stream,
                           P2, (const float*)P3, N_ACC, wo);
        hipLaunchKernelGGL(l2norm_kernel, dim3((N_MER + 7) / 8), dim3(256), 0, stream,
                           P1, (const float*)nullptr, N_MER, wo);

        // BatchNorm (training-mode batch stats) + ReLU (layers 0,1)
        hipMemsetAsync(ST, 0, 512 * sizeof(float), stream);
        hipLaunchKernelGGL(bn_stats_kernel, dim3(1024), dim3(256), 0, stream,
                           P2, N_ACC, sh, ST, ST + 128);
        hipLaunchKernelGGL(bn_stats_kernel, dim3(1024), dim3(256), 0, stream,
                           P1, N_MER, sh, ST + 256, ST + 384);
        const int relu = (l < 2) ? 1 : 0;
        hipLaunchKernelGGL(bn_apply_kernel, dim3((N_ACC * wo + 255) / 256), dim3(256), 0, stream,
                           P2, XA, ST, ST + 128, bng[l] + 0, bnb[l] + 0,
                           N_ACC * wo, sh, 1.0f / (float)N_ACC, relu);
        hipLaunchKernelGGL(bn_apply_kernel, dim3((N_MER * wo + 255) / 256), dim3(256), 0, stream,
                           P1, XM, ST + 256, ST + 384, bng[l] + wo, bnb[l] + wo,
                           N_MER * wo, sh, 1.0f / (float)N_MER, relu);
    }

    // ---- classifier head: h = relu(XA @ clfW1^T + clfb1) ; logits = h @ clfW2^T + b2 ----
    gemm(XA, Pclf1, nullptr, nullptr, clfb1, P1, N_ACC, CLFD, OUTD, 1);
    hipLaunchKernelGGL(logits_kernel, dim3((N_ACC + 7) / 8), dim3(256), 0, stream,
                       P1, clfW2, clfb2, (float*)d_out, N_ACC);
}
